// DatacubeLLMBridge_30236569764098
// MI455X (gfx1250) — matrix operand order repaired and checked by reading, not compiled, not run on hardware
//
#include <hip/hip_runtime.h>
#include <math.h>

// ---------------------------------------------------------------------------
// DatacubeLLMBridge for MI455X (gfx1250).
// All dense math runs on v_wmma_f32_16x16x32_bf16 (wave32 WMMA).
// GEMM stages tiles with CDNA5 async global->LDS DMA (ASYNCcnt) and a
// double-buffered LDS pipeline. Attention is fused flash-style.
// Physics row-biases cancel under softmax (row-constant shifts) and are omitted.
// ---------------------------------------------------------------------------

typedef __bf16 bf16;
typedef __attribute__((ext_vector_type(16))) __bf16 bf16x16;
typedef __attribute__((ext_vector_type(8)))  float   f32x8;
typedef __attribute__((ext_vector_type(4)))  float   f32x4;

union FragU { f32x4 f4[2]; bf16x16 bv; };

__device__ __forceinline__ float gelu_exact(float x) {
  return 0.5f * x * (1.0f + erff(x * 0.70710678118654752f));
}

// CDNA5 async global->LDS copy of 16 bytes (GLOBAL_LOAD_ASYNC_TO_LDS_B128).
// LDS dest address: generic shared pointers are {aperture, lds_offset} so the
// low 32 bits are the LDS byte address the VDST VGPR must carry.
__device__ __forceinline__ void async_copy16(const bf16* g, bf16* s) {
  unsigned lds = (unsigned)(unsigned long long)s;
  unsigned long long ga = (unsigned long long)g;
  asm volatile("global_load_async_to_lds_b128 %0, %1, off"
               :: "v"(lds), "v"(ga) : "memory");
}
__device__ __forceinline__ void wait_async0() {
  asm volatile("s_wait_asynccnt 0x0" ::: "memory");
}

// ---------------------------------------------------------------------------
// f32 -> bf16 conversion
// ---------------------------------------------------------------------------
__global__ void k_f32_to_bf16(const float* __restrict__ in, bf16* __restrict__ out, long n) {
  long i = (long)blockIdx.x * blockDim.x + threadIdx.x;
  if (i < n) out[i] = (bf16)in[i];
}

// ---------------------------------------------------------------------------
// Tiled WMMA GEMM:  C[M,N] = epilogue(A[M,K] @ W[N,K]^T + bias)
//   block = 256 threads (8 waves), tile 128x128, K-step 32.
//   wave (wr in 0..3, wc in 0..1) owns a 32x64 area = 2x4 WMMA tiles.
//   Tiles staged by async global->LDS DMA into a 2-deep LDS pipeline.
//   Epilogue: +bias, optional exact GELU, optional f32 residual,
//             optional f32 and/or bf16 outputs.
// Requires: M%128==0, N%128==0, K%32==0 (true for every call here).
// ---------------------------------------------------------------------------
#define BM 128
#define BN 128
#define BK 32

__global__ __launch_bounds__(256)
void k_gemm(const bf16* __restrict__ A, const bf16* __restrict__ W,
            const float* __restrict__ bias, const float* __restrict__ resid,
            float* __restrict__ outF, bf16* __restrict__ outB,
            int M, int N, int K, int gelu) {
  __shared__ __align__(16) bf16 sA[2][BM][BK];   // 16 KB
  __shared__ __align__(16) bf16 sB[2][BN][BK];   // 16 KB
  const int tid  = threadIdx.x;
  const int lane = tid & 31;
  const int wid  = tid >> 5;
  const int wr   = wid >> 1;       // 0..3
  const int wc   = wid & 1;        // 0..1
  const int l15  = lane & 15;
  const int lh   = lane >> 4;      // 0/1
  const int m0   = blockIdx.y * BM;
  const int n0   = blockIdx.x * BN;

  // this thread's two staging chunks (512 16B-chunks per tile, 2 per thread)
  const int cA = tid * 2;
  const int rowA = cA >> 2, segA = cA & 3;        // chunk 0: row, 8-elem segment
  const int rowB = (cA + 1) >> 2, segB = (cA + 1) & 3;

  // issue one tile of async copies into LDS buffer `buf`
  auto stage = [&](int buf, int k0) {
    async_copy16(&A[(size_t)(m0 + rowA) * K + k0 + segA * 8], &sA[buf][rowA][segA * 8]);
    async_copy16(&A[(size_t)(m0 + rowB) * K + k0 + segB * 8], &sA[buf][rowB][segB * 8]);
    async_copy16(&W[(size_t)(n0 + rowA) * K + k0 + segA * 8], &sB[buf][rowA][segA * 8]);
    async_copy16(&W[(size_t)(n0 + rowB) * K + k0 + segB * 8], &sB[buf][rowB][segB * 8]);
  };

  f32x8 acc[2][4];
#pragma unroll
  for (int i = 0; i < 2; ++i)
#pragma unroll
    for (int j = 0; j < 4; ++j)
#pragma unroll
      for (int v = 0; v < 8; ++v) acc[i][j][v] = 0.0f;

  stage(0, 0);
  wait_async0();
  __syncthreads();

  int cur = 0;
  for (int k0 = 0; k0 < K; k0 += BK) {
    // prefetch next tile into the alternate buffer (async DMA overlaps WMMAs)
    if (k0 + BK < K) stage(cur ^ 1, k0 + BK);

    // A fragments (16x32 bf16: lanes<16 hold K 0..15, lanes>=16 hold K 16..31)
    FragU af[2];
#pragma unroll
    for (int i = 0; i < 2; ++i) {
      int r = wr * 32 + i * 16 + l15;
      af[i].f4[0] = *(const f32x4*)&sA[cur][r][lh * 16];
      af[i].f4[1] = *(const f32x4*)&sA[cur][r][lh * 16 + 8];
    }
    // B fragments (32x16 bf16: lanes<16 hold K 0..15, lanes>=16 hold K 16..31)
#pragma unroll
    for (int j = 0; j < 4; ++j) {
      FragU bfg;
      int cc = wc * 64 + j * 16 + l15;
      bfg.f4[0] = *(const f32x4*)&sB[cur][cc][lh * 16];
      bfg.f4[1] = *(const f32x4*)&sB[cur][cc][lh * 16 + 8];
#pragma unroll
      for (int i = 0; i < 2; ++i)
        acc[i][j] = __builtin_amdgcn_wmma_f32_16x16x32_bf16(
            false, af[i].bv, false, bfg.bv, (short)0, acc[i][j], false, false);
    }

    // next buffer ready + everyone done reading `cur` before it is overwritten
    wait_async0();
    __syncthreads();
    cur ^= 1;
  }

  // epilogue (C layout: lane holds col=l15, VGPR v holds row v + 8*lh)
#pragma unroll
  for (int i = 0; i < 2; ++i)
#pragma unroll
    for (int j = 0; j < 4; ++j) {
      int colg = n0 + wc * 64 + j * 16 + l15;
      float bcol = bias ? bias[colg] : 0.0f;
#pragma unroll
      for (int v = 0; v < 8; ++v) {
        int rowg = m0 + wr * 32 + i * 16 + v + 8 * lh;
        size_t idx = (size_t)rowg * N + colg;
        float val = acc[i][j][v] + bcol;
        if (gelu)  val = gelu_exact(val);
        if (resid) val += resid[idx];
        if (outF)  outF[idx] = val;
        if (outB)  outB[idx] = (bf16)val;
      }
    }
}

// ---------------------------------------------------------------------------
// Fused flash-style attention (non-causal, wave32 WMMA).
//   Q  : bf16 [B,H,qRows,D]
//   K  : bf16 [B,H,Skp,D]
//   Vt : bf16 [B,H,D,Skp]     (pre-transposed V)
//   out: bf16 [B,Sq,H*D]      (heads merged)
// grid = (Sq/128, H, B), block = 256 (8 waves, each owns 16 query rows).
// Keys streamed in chunks of 32; keys >= Skv are masked (-1e30).
// ---------------------------------------------------------------------------
template <int D>
__global__ __launch_bounds__(256)
void k_attn(const bf16* __restrict__ Q, const bf16* __restrict__ Kmat,
            const bf16* __restrict__ Vt, bf16* __restrict__ out,
            int qRows, int Skp, int Skv, int Sq, float scale) {
  constexpr int NC = D / 32;   // score contraction chunks
  constexpr int NT = D / 16;   // output column tiles
  __shared__ __align__(16) bf16 sP[8][16][32];   // per-wave P staging, 8 KB

  const int tid = threadIdx.x, lane = tid & 31, wid = tid >> 5;
  const int l15 = lane & 15, lh = lane >> 4;
  const int h = blockIdx.y, b = blockIdx.z, H = gridDim.y;
  const int bh = b * H + h;
  const int q0 = blockIdx.x * 128 + wid * 16;
  const size_t qB = (size_t)bh * qRows * D;
  const size_t kB = (size_t)bh * Skp * D;
  const size_t vB = (size_t)bh * D * Skp;

  // load Q A-fragments once
  FragU qf[NC];
#pragma unroll
  for (int c = 0; c < NC; ++c) {
    const bf16* p = Q + qB + (size_t)(q0 + l15) * D + c * 32;
    qf[c].f4[0] = *(const f32x4*)(p + lh * 16);
    qf[c].f4[1] = *(const f32x4*)(p + lh * 16 + 8);
  }

  float mrow[8], lrow[8];
  f32x8 o[NT];
#pragma unroll
  for (int v = 0; v < 8; ++v) { mrow[v] = -3.0e38f; lrow[v] = 0.0f; }
#pragma unroll
  for (int t = 0; t < NT; ++t)
#pragma unroll
    for (int v = 0; v < 8; ++v) o[t][v] = 0.0f;

  for (int j = 0; j < Skp; j += 32) {
    // ---- scores: two 16-key tiles via WMMA ----
    f32x8 s0, s1;
#pragma unroll
    for (int v = 0; v < 8; ++v) { s0[v] = 0.0f; s1[v] = 0.0f; }
#pragma unroll
    for (int c = 0; c < NC; ++c) {
      FragU kf0, kf1;
      const bf16* p0 = Kmat + kB + (size_t)(j + l15) * D + c * 32 + lh * 16;
      const bf16* p1 = Kmat + kB + (size_t)(j + 16 + l15) * D + c * 32 + lh * 16;
      kf0.f4[0] = *(const f32x4*)p0;      kf0.f4[1] = *(const f32x4*)(p0 + 8);
      kf1.f4[0] = *(const f32x4*)p1;      kf1.f4[1] = *(const f32x4*)(p1 + 8);
      s0 = __builtin_amdgcn_wmma_f32_16x16x32_bf16(false, qf[c].bv, false, kf0.bv, (short)0, s0, false, false);
      s1 = __builtin_amdgcn_wmma_f32_16x16x32_bf16(false, qf[c].bv, false, kf1.bv, (short)0, s1, false, false);
    }
    // scale + key mask
    bool ok0 = (j + l15) < Skv, ok1 = (j + 16 + l15) < Skv;
#pragma unroll
    for (int v = 0; v < 8; ++v) {
      s0[v] = ok0 ? s0[v] * scale : -1.0e30f;
      s1[v] = ok1 ? s1[v] * scale : -1.0e30f;
    }
    // row max over the 32-key chunk (16-lane butterfly inside each half-wave)
    float cm[8];
#pragma unroll
    for (int v = 0; v < 8; ++v) cm[v] = fmaxf(s0[v], s1[v]);
#pragma unroll
    for (int off = 1; off < 16; off <<= 1)
#pragma unroll
      for (int v = 0; v < 8; ++v) cm[v] = fmaxf(cm[v], __shfl_xor(cm[v], off, 32));
    // online softmax update
    float sc[8], mn[8];
#pragma unroll
    for (int v = 0; v < 8; ++v) {
      mn[v] = fmaxf(mrow[v], cm[v]);
      sc[v] = __expf(mrow[v] - mn[v]);
      mrow[v] = mn[v];
    }
    f32x8 p0v, p1v;
    float rs[8];
#pragma unroll
    for (int v = 0; v < 8; ++v) {
      p0v[v] = __expf(s0[v] - mn[v]);
      p1v[v] = __expf(s1[v] - mn[v]);
      rs[v] = p0v[v] + p1v[v];
    }
#pragma unroll
    for (int off = 1; off < 16; off <<= 1)
#pragma unroll
      for (int v = 0; v < 8; ++v) rs[v] += __shfl_xor(rs[v], off, 32);
#pragma unroll
    for (int v = 0; v < 8; ++v) lrow[v] = lrow[v] * sc[v] + rs[v];
#pragma unroll
    for (int t = 0; t < NT; ++t)
#pragma unroll
      for (int v = 0; v < 8; ++v) o[t][v] *= sc[v];

    // ---- C-frag -> A-frag relayout of P via per-wave LDS tile ----
#pragma unroll
    for (int v = 0; v < 8; ++v) {
      int r = v + 8 * lh;
      sP[wid][r][l15]      = (bf16)p0v[v];
      sP[wid][r][16 + l15] = (bf16)p1v[v];
    }
    asm volatile("s_wait_dscnt 0x0" ::: "memory");   // CDNA5 split wait: LDS in-order per wave
    FragU pf;
    pf.f4[0] = *(const f32x4*)&sP[wid][l15][lh * 16];
    pf.f4[1] = *(const f32x4*)&sP[wid][l15][lh * 16 + 8];

    // ---- O += P @ V ----
#pragma unroll
    for (int t = 0; t < NT; ++t) {
      FragU vf;
      const bf16* pv = Vt + vB + (size_t)(t * 16 + l15) * Skp + j + lh * 16;
      vf.f4[0] = *(const f32x4*)pv;  vf.f4[1] = *(const f32x4*)(pv + 8);
      o[t] = __builtin_amdgcn_wmma_f32_16x16x32_bf16(false, pf.bv, false, vf.bv, (short)0, o[t], false, false);
    }
  }

  // normalize + write merged-head bf16 output
  const int E = H * D;
#pragma unroll
  for (int v = 0; v < 8; ++v) {
    float inv = 1.0f / lrow[v];
    int q = q0 + v + 8 * lh;
#pragma unroll
    for (int t = 0; t < NT; ++t)
      out[(size_t)(b * Sq + q) * E + h * D + t * 16 + l15] = (bf16)(o[t][v] * inv);
  }
}

// ---------------------------------------------------------------------------
// Pack fused-QKV GEMM output (f32 [B,S,3E]) into Q/K [B,H,S,d] and Vt [B,H,d,S]
// ---------------------------------------------------------------------------
__global__ void k_pack_mha(const float* __restrict__ qkv, bf16* __restrict__ Qb,
                           bf16* __restrict__ Kb, bf16* __restrict__ Vt,
                           int Bn, int S, int E, int H, int d) {
  long i = (long)blockIdx.x * 256 + threadIdx.x;
  long total = (long)Bn * S * E;
  if (i >= total) return;
  int c = (int)(i % E);
  long r = i / E;
  int s = (int)(r % S);
  int b = (int)(r / S);
  int h = c / d, dd = c % d;
  const float* base = qkv + (size_t)(b * S + s) * (3 * E);
  size_t oq = ((size_t)(b * H + h) * S + s) * d + dd;
  Qb[oq] = (bf16)base[c];
  Kb[oq] = (bf16)base[E + c];
  Vt[((size_t)(b * H + h) * d + dd) * S + s] = (bf16)base[2 * E + c];
}

// Pack a projection output (f32 [B,Sp,E]) into [B,H,Sp,d] (or [B,H,d,Sp] if transpose)
__global__ void k_pack_phys(const float* __restrict__ in, bf16* __restrict__ outp,
                            int Bn, int Sp, int E, int H, int d, int transpose) {
  long i = (long)blockIdx.x * 256 + threadIdx.x;
  long total = (long)Bn * Sp * E;
  if (i >= total) return;
  int c = (int)(i % E);
  long r = i / E;
  int s = (int)(r % Sp);
  int b = (int)(r / Sp);
  int h = c / d, dd = c % d;
  float v = in[i];
  if (transpose) outp[((size_t)(b * H + h) * d + dd) * Sp + s] = (bf16)v;
  else           outp[((size_t)(b * H + h) * Sp + s) * d + dd] = (bf16)v;
}

// Fill phys-token rows (S..S+2) and zero padding rows (S+3..Sp-1) of both aug buffers
__global__ void k_fill_phys(bf16* __restrict__ augA, bf16* __restrict__ augB,
                            const float* __restrict__ e, const float* __restrict__ m,
                            const float* __restrict__ p, int Bn, int S, int Sp, int E) {
  long i = (long)blockIdx.x * 256 + threadIdx.x;
  int pad = Sp - S;
  long total = (long)Bn * pad * E;
  if (i >= total) return;
  int c = (int)(i % E);
  long r = i / E;
  int rr = (int)(r % pad);
  int b = (int)(r / pad);
  float v = (rr == 0) ? e[c] : (rr == 1) ? m[c] : (rr == 2) ? p[c] : 0.0f;
  size_t idx = ((size_t)b * Sp + S + rr) * E + c;
  augA[idx] = (bf16)v;
  augB[idx] = (bf16)v;
}

// ---------------------------------------------------------------------------
// LayerNorm over last dim N (block per row). Optional f32 and/or bf16 outputs;
// bf16 rows can be re-indexed into a padded [*, bfStride, N] buffer.
// ---------------------------------------------------------------------------
__global__ __launch_bounds__(256)
void k_ln(const float* __restrict__ in, const float* __restrict__ g,
          const float* __restrict__ be, float* __restrict__ outF,
          bf16* __restrict__ outB, int N, int S, int bfStride) {
  const int row = blockIdx.x, tid = threadIdx.x;
  const float* x = in + (size_t)row * N;
  __shared__ float r1[256], r2[256];
  float s = 0.0f, s2 = 0.0f;
  for (int c = tid; c < N; c += 256) { float v = x[c]; s += v; s2 += v * v; }
  r1[tid] = s; r2[tid] = s2;
  __syncthreads();
  for (int o = 128; o > 0; o >>= 1) {
    if (tid < o) { r1[tid] += r1[tid + o]; r2[tid] += r2[tid + o]; }
    __syncthreads();
  }
  float mean = r1[0] / N;
  float var  = r2[0] / N - mean * mean;
  float rstd = rsqrtf(var + 1e-5f);
  size_t bfRow = (size_t)(row / S) * bfStride + (row % S);
  for (int c = tid; c < N; c += 256) {
    float y = (x[c] - mean) * rstd * g[c] + be[c];
    if (outF) outF[(size_t)row * N + c] = y;
    if (outB) outB[bfRow * (size_t)N + c] = (bf16)y;
  }
}

// ---------------------------------------------------------------------------
// Host orchestration
// ---------------------------------------------------------------------------
extern "C" void kernel_launch(void* const* d_in, const int* in_sizes, int n_in,
                              void* d_out, int out_size, void* d_ws, size_t ws_size,
                              hipStream_t stream) {
  (void)in_sizes; (void)n_in; (void)out_size; (void)ws_size;
  const int B = 4, S = 1024, E = 512, L = 768, Dbr = 1024, H = 16, dm = 32, dp = 64;
  const int Sp = 1056, Skv = 1027;         // padded / valid key counts for phys attn
  const int M  = B * S;                    // 4096
  const int Mp = B * Sp;                   // 4224

  const float* llm     = (const float*)d_in[1];
  const float* cnn0    = (const float*)d_in[2];
  const float* sa_wqkv = (const float*)d_in[3];  const float* sa_bqkv = (const float*)d_in[4];
  const float* sa_wo   = (const float*)d_in[5];  const float* sa_bo   = (const float*)d_in[6];
  const float* ta_wqkv = (const float*)d_in[7];  const float* ta_bqkv = (const float*)d_in[8];
  const float* ta_wo   = (const float*)d_in[9];  const float* ta_bo   = (const float*)d_in[10];
  const float* c2b_w = (const float*)d_in[11]; const float* c2b_b  = (const float*)d_in[12];
  const float* c2b_g = (const float*)d_in[13]; const float* c2b_be = (const float*)d_in[14];
  const float* l2b_w = (const float*)d_in[15]; const float* l2b_b  = (const float*)d_in[16];
  const float* l2b_g = (const float*)d_in[17]; const float* l2b_be = (const float*)d_in[18];
  const float* pq_w = (const float*)d_in[19]; const float* pq_b = (const float*)d_in[20];
  const float* pk_w = (const float*)d_in[21]; const float* pk_b = (const float*)d_in[22];
  const float* pv_w = (const float*)d_in[23]; const float* pv_b = (const float*)d_in[24];
  const float* po_w = (const float*)d_in[25]; const float* po_b = (const float*)d_in[26];
  const float* e_emb = (const float*)d_in[27]; const float* m_emb = (const float*)d_in[28];
  const float* p_emb = (const float*)d_in[29];
  const float* pa_g = (const float*)d_in[30]; const float* pa_be = (const float*)d_in[31];
  const float* c2l_w = (const float*)d_in[32]; const float* c2l_b  = (const float*)d_in[33];
  const float* c2l_g = (const float*)d_in[34]; const float* c2l_be = (const float*)d_in[35];
  const float* l2c_w = (const float*)d_in[36]; const float* l2c_b  = (const float*)d_in[37];
  const float* l2c_g = (const float*)d_in[38]; const float* l2c_be = (const float*)d_in[39];

  float* out_cnn = (float*)d_out;                       // [4096,512]
  float* out_llm = (float*)d_out + (size_t)M * E;       // [4096,768]

  size_t off = 0;
  auto alloc  = [&](size_t bytes) -> void* {
    void* p = (char*)d_ws + off;
    off += (bytes + 255) & ~(size_t)255;
    return p;
  };
  auto allocB = [&](size_t n) -> bf16*  { return (bf16*)alloc(n * sizeof(bf16)); };
  auto allocF = [&](size_t n) -> float* { return (float*)alloc(n * sizeof(float)); };

  auto conv = [&](const float* src, bf16* dst, long n) {
    k_f32_to_bf16<<<(unsigned)((n + 255) / 256), 256, 0, stream>>>(src, dst, n);
  };
  auto gemm = [&](const bf16* A, const bf16* W, const float* bias, const float* resid,
                  float* oF, bf16* oB, int Mm, int Nn, int Kk, int gelu) {
    dim3 g(Nn / 128, Mm / 128);
    k_gemm<<<g, 256, 0, stream>>>(A, W, bias, resid, oF, oB, Mm, Nn, Kk, gelu);
  };

  // ---- bf16 weight copies ----
  bf16* w_saqkv = allocB((size_t)3 * E * E);
  bf16* w_sao   = allocB((size_t)E * E);
  bf16* w_taqkv = allocB((size_t)3 * E * E);
  bf16* w_tao   = allocB((size_t)E * E);
  bf16* w_c2b   = allocB((size_t)Dbr * E);
  bf16* w_l2b   = allocB((size_t)Dbr * L);
  bf16* w_pq    = allocB((size_t)Dbr * Dbr);
  bf16* w_pk    = allocB((size_t)Dbr * Dbr);
  bf16* w_pv    = allocB((size_t)Dbr * Dbr);
  bf16* w_po    = allocB((size_t)Dbr * Dbr);
  bf16* w_c2l   = allocB((size_t)L * Dbr);
  bf16* w_l2c   = allocB((size_t)E * Dbr);
  conv(sa_wqkv, w_saqkv, (long)3 * E * E);  conv(sa_wo, w_sao, (long)E * E);
  conv(ta_wqkv, w_taqkv, (long)3 * E * E);  conv(ta_wo, w_tao, (long)E * E);
  conv(c2b_w, w_c2b, (long)Dbr * E);        conv(l2b_w, w_l2b, (long)Dbr * L);
  conv(pq_w, w_pq, (long)Dbr * Dbr);        conv(pk_w, w_pk, (long)Dbr * Dbr);
  conv(pv_w, w_pv, (long)Dbr * Dbr);        conv(po_w, w_po, (long)Dbr * Dbr);
  conv(c2l_w, w_c2l, (long)L * Dbr);        conv(l2c_w, w_l2c, (long)E * Dbr);

  // ---- activation buffers ----
  bf16*  x0_bf   = allocB((size_t)M * E);
  bf16*  llm_bf  = allocB((size_t)M * L);
  float* qkv_f   = allocF((size_t)M * 3 * E);
  bf16*  Q1      = allocB((size_t)B * H * S * dm);
  bf16*  K1      = allocB((size_t)B * H * S * dm);
  bf16*  V1t     = allocB((size_t)B * H * dm * S);
  bf16*  ao_bf   = allocB((size_t)M * E);
  float* cnn1_f  = allocF((size_t)M * E);
  bf16*  cnn1_bf = allocB((size_t)M * E);
  float* cnn2_f  = allocF((size_t)M * E);
  bf16*  cnn2_bf = allocB((size_t)M * E);
  float* t1_f    = allocF((size_t)M * Dbr);
  float* cnnb_f  = allocF((size_t)M * Dbr);
  float* llmb_f  = allocF((size_t)M * Dbr);
  bf16*  aug_cnn = allocB((size_t)Mp * Dbr);
  bf16*  aug_llm = allocB((size_t)Mp * Dbr);
  float* scr_f   = allocF((size_t)Mp * Dbr);
  bf16*  Qp      = allocB((size_t)B * H * Sp * dp);
  bf16*  Kp      = allocB((size_t)B * H * Sp * dp);
  bf16*  Vtp     = allocB((size_t)B * H * dp * Sp);
  bf16*  op_bf   = allocB((size_t)M * Dbr);
  float* t3_f    = allocF((size_t)M * Dbr);
  bf16*  catt_bf = allocB((size_t)M * Dbr);
  bf16*  latt_bf = allocB((size_t)M * Dbr);
  float* u_f     = allocF((size_t)M * L);

  const float scale_m = 0.17677669529663687f;  // 1/sqrt(32)
  const float scale_p = 0.125f;                // 1/sqrt(64)

  // ---- CNN MHA block 1 (spatial) ----
  conv(cnn0, x0_bf, (long)M * E);
  gemm(x0_bf, w_saqkv, sa_bqkv, nullptr, qkv_f, nullptr, M, 3 * E, E, 0);
  k_pack_mha<<<(unsigned)(((long)M * E + 255) / 256), 256, 0, stream>>>(qkv_f, Q1, K1, V1t, B, S, E, H, dm);
  k_attn<32><<<dim3(S / 128, H, B), 256, 0, stream>>>(Q1, K1, V1t, ao_bf, S, S, S, S, scale_m);
  gemm(ao_bf, w_sao, sa_bo, cnn0, cnn1_f, cnn1_bf, M, E, E, 0);

  // ---- CNN MHA block 2 (temporal) ----
  gemm(cnn1_bf, w_taqkv, ta_bqkv, nullptr, qkv_f, nullptr, M, 3 * E, E, 0);
  k_pack_mha<<<(unsigned)(((long)M * E + 255) / 256), 256, 0, stream>>>(qkv_f, Q1, K1, V1t, B, S, E, H, dm);
  k_attn<32><<<dim3(S / 128, H, B), 256, 0, stream>>>(Q1, K1, V1t, ao_bf, S, S, S, S, scale_m);
  gemm(ao_bf, w_tao, ta_bo, cnn1_f, cnn2_f, cnn2_bf, M, E, E, 0);

  // ---- bridges to BR_D + LayerNorms (bf16 rows land in padded aug buffers) ----
  gemm(cnn2_bf, w_c2b, c2b_b, nullptr, t1_f, nullptr, M, Dbr, E, 1);
  k_ln<<<M, 256, 0, stream>>>(t1_f, c2b_g, c2b_be, cnnb_f, aug_cnn, Dbr, S, Sp);
  conv(llm, llm_bf, (long)M * L);
  gemm(llm_bf, w_l2b, l2b_b, nullptr, t1_f, nullptr, M, Dbr, L, 1);
  k_ln<<<M, 256, 0, stream>>>(t1_f, l2b_g, l2b_be, llmb_f, aug_llm, Dbr, S, Sp);
  k_fill_phys<<<(unsigned)(((long)B * (Sp - S) * Dbr + 255) / 256), 256, 0, stream>>>(
      aug_cnn, aug_llm, e_emb, m_emb, p_emb, B, S, Sp, Dbr);

  const long packN = (long)Mp * Dbr;
  // ---- physics cross-attention 1: queries = aug_cnn, keys = aug_llm ----
  gemm(aug_cnn, w_pq, pq_b, nullptr, scr_f, nullptr, Mp, Dbr, Dbr, 0);
  k_pack_phys<<<(unsigned)((packN + 255) / 256), 256, 0, stream>>>(scr_f, Qp, B, Sp, Dbr, H, dp, 0);
  gemm(aug_llm, w_pk, pk_b, nullptr, scr_f, nullptr, Mp, Dbr, Dbr, 0);
  k_pack_phys<<<(unsigned)((packN + 255) / 256), 256, 0, stream>>>(scr_f, Kp, B, Sp, Dbr, H, dp, 0);
  gemm(aug_llm, w_pv, pv_b, nullptr, scr_f, nullptr, Mp, Dbr, Dbr, 0);
  k_pack_phys<<<(unsigned)((packN + 255) / 256), 256, 0, stream>>>(scr_f, Vtp, B, Sp, Dbr, H, dp, 1);
  k_attn<64><<<dim3(S / 128, H, B), 256, 0, stream>>>(Qp, Kp, Vtp, op_bf, Sp, Sp, Skv, S, scale_p);
  gemm(op_bf, w_po, po_b, cnnb_f, t3_f, nullptr, M, Dbr, Dbr, 0);
  k_ln<<<M, 256, 0, stream>>>(t3_f, pa_g, pa_be, nullptr, catt_bf, Dbr, S, S);

  // ---- physics cross-attention 2: queries = aug_llm, keys = aug_cnn ----
  gemm(aug_llm, w_pq, pq_b, nullptr, scr_f, nullptr, Mp, Dbr, Dbr, 0);
  k_pack_phys<<<(unsigned)((packN + 255) / 256), 256, 0, stream>>>(scr_f, Qp, B, Sp, Dbr, H, dp, 0);
  gemm(aug_cnn, w_pk, pk_b, nullptr, scr_f, nullptr, Mp, Dbr, Dbr, 0);
  k_pack_phys<<<(unsigned)((packN + 255) / 256), 256, 0, stream>>>(scr_f, Kp, B, Sp, Dbr, H, dp, 0);
  gemm(aug_cnn, w_pv, pv_b, nullptr, scr_f, nullptr, Mp, Dbr, Dbr, 0);
  k_pack_phys<<<(unsigned)((packN + 255) / 256), 256, 0, stream>>>(scr_f, Vtp, B, Sp, Dbr, H, dp, 1);
  k_attn<64><<<dim3(S / 128, H, B), 256, 0, stream>>>(Qp, Kp, Vtp, op_bf, Sp, Sp, Skv, S, scale_p);
  gemm(op_bf, w_po, po_b, llmb_f, t3_f, nullptr, M, Dbr, Dbr, 0);
  k_ln<<<M, 256, 0, stream>>>(t3_f, pa_g, pa_be, nullptr, latt_bf, Dbr, S, S);

  // ---- final bridges ----
  gemm(catt_bf, w_c2l, c2l_b, nullptr, u_f, nullptr, M, L, Dbr, 1);
  k_ln<<<M, 256, 0, stream>>>(u_f, c2l_g, c2l_be, out_llm, nullptr, L, S, S);
  gemm(latt_bf, w_l2c, l2c_b, nullptr, u_f, nullptr, M, E, Dbr, 1);
  k_ln<<<M, 256, 0, stream>>>(u_f, l2c_g, l2c_be, out_cnn, nullptr, E, S, S);
}
